// MyHypergraphConv_23089744183324
// MI455X (gfx1250) — compile-verified, matrix operand order checked by reading, not run on hardware
//
#include <hip/hip_runtime.h>

// ---------------------------------------------------------------------------
// MyHypergraphConv for MI455X (gfx1250, wave32, WMMA bf16)
//
//   deg_n = sum(H,1); deg_e = sum(H,0); dn = rsqrt(deg_n); de = 1/deg_e
//   eT  = (xsT @ H) * de      via  A = xsT,      BT = HbT   (bf16 WMMA)
//   xn  = (H @ e) * dn        via  A = Hb,       BT = eT    (bf16 WMMA)
//   out = xn @ W^T + b        via  A = xn(bf16), BT = W     (bf16 WMMA)
//
// All WMMA fragment loads are contiguous 16B global loads (A rows and BT rows
// are K-contiguous by construction), so the GEMM kernels need no LDS staging.
// ---------------------------------------------------------------------------

#define N_NODES 8192
#define N_EDGES 4096
#define D_IN    256
#define D_OUT   256

typedef __attribute__((ext_vector_type(16))) __bf16 v16bf;
typedef __attribute__((ext_vector_type(8)))  __bf16 v8bf;
typedef __attribute__((ext_vector_type(8)))  float  v8f;

// ---------------------------------------------------------------------------
// Degree reductions
// ---------------------------------------------------------------------------
__global__ void row_deg_kernel(const float* __restrict__ H, float* __restrict__ dn,
                               int R, int C) {
    __shared__ float red[256];
    const int row = blockIdx.x;
    const int tid = threadIdx.x;
    float s = 0.f;
    const float* rp = H + (size_t)row * C;
    for (int c = tid; c < C; c += 256) s += rp[c];
    red[tid] = s;
    __syncthreads();
    for (int off = 128; off > 0; off >>= 1) {
        if (tid < off) red[tid] += red[tid + off];
        __syncthreads();
    }
    if (tid == 0) {
        float d = red[0];
        dn[row] = (d > 0.f) ? rsqrtf(d) : 0.f;
    }
}

__global__ void col_deg_kernel(const float* __restrict__ H, float* __restrict__ de,
                               int R, int C) {
    const int j = blockIdx.x * blockDim.x + threadIdx.x;
    if (j >= C) return;
    float s = 0.f;
    for (int i = 0; i < R; ++i) s += H[(size_t)i * C + j];
    de[j] = (s > 0.f) ? (1.f / s) : 0.f;
}

// ---------------------------------------------------------------------------
// Conversions / transposes (32x32 LDS-tiled so every global access is
// lane-contiguous; block = 32x8 threads, 4 rows per thread)
// ---------------------------------------------------------------------------
// Hb[i][j] = bf16(H[i][j]);  HbT[j][i] = bf16(H[i][j])
__global__ void conv_h_kernel(const float* __restrict__ H,
                              __bf16* __restrict__ Hb, __bf16* __restrict__ HbT) {
    __shared__ __bf16 t[32][33];
    const int j0 = blockIdx.x * 32;
    const int i0 = blockIdx.y * 32;
    const int tx = threadIdx.x;   // 0..31
    const int ty = threadIdx.y;   // 0..7
#pragma unroll
    for (int k = 0; k < 4; ++k) {
        const int i = i0 + ty + 8 * k;
        const size_t idx = (size_t)i * N_EDGES + j0 + tx;
        const __bf16 bv = (__bf16)H[idx];
        Hb[idx] = bv;
        t[ty + 8 * k][tx] = bv;
    }
    __syncthreads();
#pragma unroll
    for (int k = 0; k < 4; ++k) {
        const int j = j0 + ty + 8 * k;
        HbT[(size_t)j * N_NODES + i0 + tx] = t[tx][ty + 8 * k];
    }
}

// xsT[d][i] = bf16(x[i][d] * dn[i])
__global__ void build_xst_kernel(const float* __restrict__ x, const float* __restrict__ dn,
                                 __bf16* __restrict__ xsT) {
    __shared__ __bf16 t[32][33];
    const int d0 = blockIdx.y * 32;
    const int i0 = blockIdx.x * 32;
    const int tx = threadIdx.x;
    const int ty = threadIdx.y;
#pragma unroll
    for (int k = 0; k < 4; ++k) {
        const int i = i0 + ty + 8 * k;
        const float v = x[(size_t)i * D_IN + d0 + tx] * dn[i];
        t[ty + 8 * k][tx] = (__bf16)v;
    }
    __syncthreads();
#pragma unroll
    for (int k = 0; k < 4; ++k) {
        const int d = d0 + ty + 8 * k;
        xsT[(size_t)d * N_NODES + i0 + tx] = t[tx][ty + 8 * k];
    }
}

__global__ void conv_w_kernel(const float* __restrict__ W, __bf16* __restrict__ Wb) {
    const int idx = blockIdx.x * blockDim.x + threadIdx.x;
    Wb[idx] = (__bf16)W[idx];
}

// ---------------------------------------------------------------------------
// bf16 WMMA GEMM:  C[M,N] = A[M,K] * BT[N,K]^T, optional diag scales / bias.
// Block: 256 threads = 8 waves (4 M x 2 N); each wave owns a 64x64 tile
// (4x4 v_wmma_f32_16x16x32_bf16 tiles, 128 accumulator VGPRs).
// ---------------------------------------------------------------------------
__device__ __forceinline__ v16bf load_frag(const __bf16* base) {
    // 16-bit A/B fragment: lanes 0-15 hold K {0..7,16..23}, lanes 16-31 hold
    // K {8..15,24..31}; khalf is folded into `base` by the caller.
    v8bf lo = *(const v8bf*)(base);
    v8bf hi = *(const v8bf*)(base + 16);
    return __builtin_shufflevector(lo, hi, 0, 1, 2, 3, 4, 5, 6, 7,
                                           8, 9, 10, 11, 12, 13, 14, 15);
}

__global__ void __launch_bounds__(256)
gemm_wmma_kernel(const __bf16* __restrict__ A, int lda,
                 const __bf16* __restrict__ BT, int ldb, int K,
                 const float* __restrict__ scaleM,   // per-row scale or null
                 const float* __restrict__ scaleN,   // per-col scale or null
                 const float* __restrict__ bias,     // per-col bias or null
                 float* __restrict__ outF,           // f32 output or null
                 __bf16* __restrict__ outB,          // bf16 output or null
                 int ldo) {
    const int lane  = threadIdx.x & 31;
    const int wave  = threadIdx.x >> 5;            // 0..7
    const int r     = lane & 15;                   // fragment row within tile
    const int half  = lane >> 4;                   // 0 / 1
    const int khalf = half * 8;                    // ISA 16-bit lane split

    const int m0 = blockIdx.x * 256 + (wave >> 1) * 64;   // wave M origin
    const int n0 = blockIdx.y * 128 + (wave & 1) * 64;    // wave N origin

    const __bf16* aRow[4];
    const __bf16* bRow[4];
#pragma unroll
    for (int t = 0; t < 4; ++t) {
        aRow[t] = A  + (size_t)(m0 + t * 16 + r) * lda + khalf;
        bRow[t] = BT + (size_t)(n0 + t * 16 + r) * ldb + khalf;
    }

    v8f acc[4][4];
#pragma unroll
    for (int i = 0; i < 4; ++i)
#pragma unroll
        for (int j = 0; j < 4; ++j)
            acc[i][j] = (v8f){0.f, 0.f, 0.f, 0.f, 0.f, 0.f, 0.f, 0.f};

    for (int kk = 0; kk < K; kk += 32) {
        if (kk + 128 < K) {  // stream next tiles into near caches
            __builtin_prefetch(aRow[0] + kk + 128, 0, 3);
            __builtin_prefetch(bRow[0] + kk + 128, 0, 3);
        }
        v16bf af[4], bf[4];
#pragma unroll
        for (int t = 0; t < 4; ++t) af[t] = load_frag(aRow[t] + kk);
#pragma unroll
        for (int t = 0; t < 4; ++t) bf[t] = load_frag(bRow[t] + kk);
#pragma unroll
        for (int tm = 0; tm < 4; ++tm)
#pragma unroll
            for (int tn = 0; tn < 4; ++tn)
                acc[tm][tn] = __builtin_amdgcn_wmma_f32_16x16x32_bf16(
                    false, af[tm], false, bf[tn], (short)0, acc[tm][tn],
                    false, false);
    }

    // Epilogue. C layout: VGPR i, lanes0-15 -> (M=m0+i, N=lane); lanes16-31 ->
    // (M=m0+8+i, N=lane-16).  v = (scaleM? v*dn[m] : v) * sn + bn covers all
    // three launch modes (sn defaults 1, bn defaults 0).
#pragma unroll
    for (int tm = 0; tm < 4; ++tm) {
#pragma unroll
        for (int tn = 0; tn < 4; ++tn) {
            const int n = n0 + tn * 16 + r;
            const float sn = scaleN ? scaleN[n] : 1.f;
            const float bn = bias ? bias[n] : 0.f;
#pragma unroll
            for (int i = 0; i < 8; ++i) {
                const int m = m0 + tm * 16 + i + half * 8;
                float v = acc[tm][tn][i];
                if (scaleM) v *= scaleM[m];
                v = v * sn + bn;
                if (outF) outF[(size_t)m * ldo + n] = v;
                else      outB[(size_t)m * ldo + n] = (__bf16)v;
            }
        }
    }
}

// ---------------------------------------------------------------------------
// Launch
// ---------------------------------------------------------------------------
extern "C" void kernel_launch(void* const* d_in, const int* in_sizes, int n_in,
                              void* d_out, int out_size, void* d_ws, size_t ws_size,
                              hipStream_t stream) {
    const float* x = (const float*)d_in[0];   // [8192,256]
    const float* H = (const float*)d_in[1];   // [8192,4096]
    const float* W = (const float*)d_in[2];   // [256,256]
    const float* b = (const float*)d_in[3];   // [256]
    float* out = (float*)d_out;               // [8192,256]

    // Workspace partition (all offsets 256B aligned).
    char* ws = (char*)d_ws;
    __bf16* Hb  = (__bf16*)(ws);                               // 64 MB
    __bf16* HbT = (__bf16*)(ws + (size_t)67108864);            // 64 MB
    __bf16* xsT = (__bf16*)(ws + (size_t)134217728);           // 4 MB
    __bf16* eT  = (__bf16*)(ws + (size_t)138412032);           // 2 MB
    __bf16* xnb = (__bf16*)(ws + (size_t)140509184);           // 4 MB
    __bf16* Wb  = (__bf16*)(ws + (size_t)144703488);           // 128 KB
    float*  dn  = (float*)(ws + (size_t)144834560);            // 32 KB
    float*  de  = (float*)(ws + (size_t)144867328);            // 16 KB

    // 1) degrees / scalings
    row_deg_kernel<<<N_NODES, 256, 0, stream>>>(H, dn, N_NODES, N_EDGES);
    col_deg_kernel<<<N_EDGES / 256, 256, 0, stream>>>(H, de, N_NODES, N_EDGES);

    // 2) precision conversions + tiled transposes
    conv_h_kernel<<<dim3(N_EDGES / 32, N_NODES / 32), dim3(32, 8), 0, stream>>>(H, Hb, HbT);
    build_xst_kernel<<<dim3(N_NODES / 32, D_IN / 32), dim3(32, 8), 0, stream>>>(x, dn, xsT);
    conv_w_kernel<<<(D_OUT * D_IN) / 256, 256, 0, stream>>>(W, Wb);

    // 3) eT[256,4096] = (xsT @ H) * de   : A=xsT, BT=HbT, K=8192
    gemm_wmma_kernel<<<dim3(D_IN / 256, N_EDGES / 128), 256, 0, stream>>>(
        xsT, N_NODES, HbT, N_NODES, N_NODES,
        nullptr, de, nullptr, nullptr, eT, N_EDGES);

    // 4) xn[8192,256] = (H @ e) * dn     : A=Hb, BT=eT, K=4096
    gemm_wmma_kernel<<<dim3(N_NODES / 256, D_IN / 128), 256, 0, stream>>>(
        Hb, N_EDGES, eT, N_EDGES, N_EDGES,
        dn, nullptr, nullptr, nullptr, xnb, D_IN);

    // 5) out[8192,256] = xn @ W^T + b    : A=xnb, BT=Wb, K=256
    gemm_wmma_kernel<<<dim3(N_NODES / 256, D_OUT / 128), 256, 0, stream>>>(
        xnb, D_IN, Wb, D_IN, D_IN,
        nullptr, nullptr, b, out, nullptr, D_OUT);
}
